// WindowAttention_10067403342465
// MI455X (gfx1250) — matrix operand order repaired and checked by reading, not compiled, not run on hardware
//
#include <hip/hip_runtime.h>
#include <hip/hip_bf16.h>

typedef __attribute__((ext_vector_type(16))) _Float16 v16h;
typedef __attribute__((ext_vector_type(8)))  _Float16 v8h;
typedef __attribute__((ext_vector_type(8)))  float    v8f;

#define DIM   192
#define HEADS 6
#define HD    32
#define NTOK  64

// LDS row strides (halves): multiples of 8 halves (16B) for aligned v8h ds
// ops; stride-in-dwords chosen so 16 lanes reading consecutive rows spread
// across banks.
#define QP  40   // sQ  [6][64][QP]   (token-major per head)
#define VP  40   // sV  [6][64][VP]
#define KP  72   // sKT [6][32][KP]   (K transposed: channel-major)
#define XTP 72   // sXT [192][XTP]    (X^T during stage 1)
#define OP  200  // sO  [64][OP]      (aliases sXT region after stage 1)
#define PP  72   // sP  [8][16][PP]   per-wave softmax rows

#define SQ_OFF   0
#define SV_OFF   (SQ_OFF  + 6*64*QP)       // 15360
#define SKT_OFF  (SV_OFF  + 6*64*VP)       // 30720
#define SXO_OFF  (SKT_OFF + 6*32*KP)       // 44544 (X^T: 13824 h / O: 12800 h)
#define SP_OFF   (SXO_OFF + 192*XTP)       // 58368
#define SMEM_HALVES (SP_OFF + 8*16*PP)     // 67584 halves = 135168 B (<320KB/WG)

// Workspace layout (bytes)
#define WS_WQKVH  0                         // f16 [576][192] natural = 221184 B
#define WS_WPROJT 221184                    // f16 [192][192] K-major =  73728 B
#define WS_BIASF  (221184 + 73728)          // f32 [6][64][64]         =  98304 B

static __device__ inline v16h cat16(v8h lo, v8h hi) {
  v16h r;
#pragma unroll
  for (int i = 0; i < 8; ++i) { r[i] = lo[i]; r[8 + i] = hi[i]; }
  return r;
}

static __device__ inline v8f wmma32(v16h a, v16h b, v8f c) {
  return __builtin_amdgcn_wmma_f32_16x16x32_f16(false, a, false, b,
                                                (short)0, c, false, false);
}

// ---------------------------------------------------------------------------
// Prep: f32->f16 weight casts + relative-position bias gather (all L2-resident)
// ---------------------------------------------------------------------------
__global__ void winattn_prep_kernel(const float* __restrict__ qkv_w,
                                    const float* __restrict__ proj_w,
                                    const float* __restrict__ bias_table,
                                    const int*   __restrict__ rel_idx,
                                    _Float16* __restrict__ wqkvH,
                                    _Float16* __restrict__ wprojT,
                                    float* __restrict__ biasF) {
  const int stride = gridDim.x * blockDim.x;
  const int t0 = blockIdx.x * blockDim.x + threadIdx.x;
  for (int o = t0; o < 576 * 192; o += stride)          // natural [n][k]
    wqkvH[o] = (_Float16)qkv_w[o];
  for (int o = t0; o < 192 * 192; o += stride) {        // wprojT[k][n]
    int k = o / 192, n = o % 192;
    wprojT[o] = (_Float16)proj_w[n * 192 + k];
  }
  for (int o = t0; o < HEADS * 64 * 64; o += stride) {  // biasF[h][q][k]
    int h = o / 4096, qk = o % 4096;
    biasF[o] = bias_table[rel_idx[qk] * HEADS + h];
  }
}

// ---------------------------------------------------------------------------
// Fused window attention: one block (8 wave32) per window.
// ---------------------------------------------------------------------------
__global__ __launch_bounds__(256) void winattn_main_kernel(
    const float* __restrict__ x, const float* __restrict__ mask,
    const float* __restrict__ qkv_b, const float* __restrict__ proj_b,
    const _Float16* __restrict__ wqkvH, const _Float16* __restrict__ wprojT,
    const float* __restrict__ biasF, float* __restrict__ out, int nW) {
  extern __shared__ _Float16 smem[];
  _Float16* sQ  = smem + SQ_OFF;
  _Float16* sV  = smem + SV_OFF;
  _Float16* sKT = smem + SKT_OFF;
  _Float16* sXT = smem + SXO_OFF;   // X^T in stage 1
  _Float16* sO  = smem + SXO_OFF;   // O afterwards (X^T is dead)
  _Float16* sP  = smem + SP_OFF;

  const int b    = blockIdx.x;
  const int tid  = threadIdx.x;
  const int wave = __builtin_amdgcn_readfirstlane(tid >> 5);
  const int lane = tid & 31;
  const int ln   = lane & 15;  // fragment row/col index
  const int grp  = lane >> 4;  // half-wave group
  const int kb   = grp * 8;    // A-fragment K sub-offset

  // ---- Stage 0: X (64x192 f32) -> transposed f16 LDS ----------------------
  {
    const float4* xv = (const float4*)(x + (size_t)b * (NTOK * DIM));
#pragma unroll
    for (int i = 0; i < 12; ++i) {   // 3072 float4 / 256 threads
      int idx = tid + i * 256;
      int e = idx * 4;
      int row = e / DIM, col = e % DIM;
      float4 v = xv[idx];
      sXT[(col + 0) * XTP + row] = (_Float16)v.x;
      sXT[(col + 1) * XTP + row] = (_Float16)v.y;
      sXT[(col + 2) * XTP + row] = (_Float16)v.z;
      sXT[(col + 3) * XTP + row] = (_Float16)v.w;
    }
  }
  __syncthreads();

  // ---- Stage 1: QKV^T = Wqkv x X^T (A = weight rows, B = X^T tile) --------
  // Wave owns a fixed 16-token column block -> the 6 B fragments are loop-
  // invariant registers; C rows are 8 consecutive channels of one token, so
  // Q/V scatters are single ds_store_b128.
  const float qscale = 0.1767766952966369f;  // hd^-0.5
  {
    const int mt1  = wave & 3;
    const int tokc = mt1 * 16 + ln;
    v16h Bf[6];
#pragma unroll
    for (int kk = 0; kk < 6; ++kk) {
      for (int s = 0; s < 16; ++s)
        Bf[kk][s] = sXT[(kk * 32 + kb + (s & 7) + ((s >> 3) << 4)) * XTP + mt1 * 16 + ln];
    }
    for (int j = 0; j < 18; ++j) {   // nt: 36 feature tiles / 2 wave-groups
      const int nt = __builtin_amdgcn_readfirstlane((wave >> 2) + 2 * j);
      if (nt + 2 < 36)  // speculative prefetch of next weight tile (L2)
        __builtin_prefetch(wqkvH + (size_t)((nt + 2) * 16 + ln) * DIM, 0, 0);
      v8f acc = {};
#pragma unroll
      for (int kk = 0; kk < 6; ++kk) {
        const _Float16* ap = wqkvH + (size_t)(nt * 16 + ln) * DIM + kk * 32 + kb;
        v16h a = cat16(*(const v8h*)ap, *(const v8h*)(ap + 16));
        acc = wmma32(a, Bf[kk], acc);
      }
      const int sub  = nt % 12;
      const int head = sub >> 1;
      const int d0b  = (sub & 1) * 16 + kb;    // first of 8 consecutive channels
      const int nb   = nt * 16 + kb;
      float4 b0 = *(const float4*)(qkv_b + nb);
      float4 b1 = *(const float4*)(qkv_b + nb + 4);
      const float bias[8] = {b0.x, b0.y, b0.z, b0.w, b1.x, b1.y, b1.z, b1.w};
      const int which = nt / 12;               // scalar branch (nt is SGPR)
      if (which == 0) {
        v8h pk;
#pragma unroll
        for (int r = 0; r < 8; ++r)
          pk[r] = (_Float16)((acc[r] + bias[r]) * qscale);
        *(v8h*)(sQ + (head * 64 + tokc) * QP + d0b) = pk;
      } else if (which == 1) {
#pragma unroll
        for (int r = 0; r < 8; ++r)            // K stored channel-major
          sKT[(head * 32 + d0b + r) * KP + tokc] = (_Float16)(acc[r] + bias[r]);
      } else {
        v8h pk;
#pragma unroll
        for (int r = 0; r < 8; ++r)
          pk[r] = (_Float16)(acc[r] + bias[r]);
        *(v8h*)(sV + (head * 64 + tokc) * VP + d0b) = pk;
      }
    }
  }
  __syncthreads();

  // ---- Stage 2: per (head, 16-row block): QK^T + softmax + PV -------------
  const int wi = b % nW;
  const float* maskw = mask + (size_t)wi * (NTOK * NTOK);
  for (int j = 0; j < 3; ++j) {
    const int u  = __builtin_amdgcn_readfirstlane(wave + 8 * j);
    const int h  = u >> 2;
    const int mt = u & 3;
    const _Float16* qp = sQ + (h * 64 + mt * 16 + ln) * QP + kb;
    const v16h a = cat16(*(const v8h*)qp, *(const v8h*)(qp + 16));
    v8f acc[4];
#pragma unroll
    for (int kt = 0; kt < 4; ++kt) {
      v16h bm;
      for (int s = 0; s < 16; ++s)
        bm[s] = sKT[(h * 32 + kb + (s & 7) + ((s >> 3) << 4)) * KP + kt * 16 + ln];
      v8f z = {};
      acc[kt] = wmma32(a, bm, z);
    }
    _Float16* prow = sP + wave * (16 * PP);
#pragma unroll
    for (int r = 0; r < 8; ++r) {
      const int qrow = mt * 16 + grp * 8 + r;
      const float* bi = biasF + (h * 4096 + qrow * 64);
      const float* mk = maskw + qrow * 64;
      float v0 = acc[0][r] + bi[ln]      + mk[ln];
      float v1 = acc[1][r] + bi[16 + ln] + mk[16 + ln];
      float v2 = acc[2][r] + bi[32 + ln] + mk[32 + ln];
      float v3 = acc[3][r] + bi[48 + ln] + mk[48 + ln];
      float m = fmaxf(fmaxf(v0, v1), fmaxf(v2, v3));
#pragma unroll
      for (int off = 1; off < 16; off <<= 1)
        m = fmaxf(m, __shfl_xor(m, off, 32));   // stays within 16-lane group
      float e0 = __expf(v0 - m), e1 = __expf(v1 - m);
      float e2 = __expf(v2 - m), e3 = __expf(v3 - m);
      float s = e0 + e1 + e2 + e3;
#pragma unroll
      for (int off = 1; off < 16; off <<= 1)
        s += __shfl_xor(s, off, 32);
      const float inv = 1.0f / s;
      _Float16* pr = prow + (grp * 8 + r) * PP;
      pr[ln]      = (_Float16)(e0 * inv);
      pr[16 + ln] = (_Float16)(e1 * inv);
      pr[32 + ln] = (_Float16)(e2 * inv);
      pr[48 + ln] = (_Float16)(e3 * inv);
    }
    // P (16x64) @ V[h] (64x32) -> O rows (O aliases dead X^T region)
#pragma unroll
    for (int nt2 = 0; nt2 < 2; ++nt2) {
      v8f acc2 = {};
#pragma unroll
      for (int kk = 0; kk < 2; ++kk) {
        const _Float16* pa = prow + ln * PP + kk * 32 + kb;
        v16h a2 = cat16(*(const v8h*)pa, *(const v8h*)(pa + 16));
        v16h b2;
        for (int s = 0; s < 16; ++s)
          b2[s] = sV[(h * 64 + kk * 32 + kb + (s & 7) + ((s >> 3) << 4)) * VP + nt2 * 16 + ln];
        acc2 = wmma32(a2, b2, acc2);
      }
#pragma unroll
      for (int r = 0; r < 8; ++r) {
        const int tok = mt * 16 + grp * 8 + r;
        sO[tok * OP + h * 32 + nt2 * 16 + ln] = (_Float16)acc2[r];
      }
    }
  }
  __syncthreads();

  // ---- Stage 3: Y = O @ WprojT + proj_b -> global f32 ---------------------
  float* op = out + (size_t)b * (NTOK * DIM);
  for (int j = 0; j < 6; ++j) {
    const int t  = __builtin_amdgcn_readfirstlane(wave + 8 * j);
    const int mt = t / 12, nt = t % 12;
    v8f acc = {};
#pragma unroll
    for (int kk = 0; kk < 6; ++kk) {
      const _Float16* ap = sO + (mt * 16 + ln) * OP + kk * 32 + kb;
      v16h a = cat16(*(const v8h*)ap, *(const v8h*)(ap + 16));
      v16h bm;
      for (int s = 0; s < 16; ++s)
        bm[s] = wprojT[(size_t)(kk * 32 + kb + (s & 7) + ((s >> 3) << 4)) * 192 + nt * 16 + ln];
      acc = wmma32(a, bm, acc);
    }
    const int col = nt * 16 + ln;
    const float pb = proj_b[col];
#pragma unroll
    for (int r = 0; r < 8; ++r) {
      const int tok = mt * 16 + grp * 8 + r;
      op[tok * DIM + col] = acc[r] + pb;
    }
  }
}

// ---------------------------------------------------------------------------
extern "C" void kernel_launch(void* const* d_in, const int* in_sizes, int n_in,
                              void* d_out, int out_size, void* d_ws, size_t ws_size,
                              hipStream_t stream) {
  const float* x      = (const float*)d_in[0];
  const float* mask   = (const float*)d_in[1];
  const float* qkv_w  = (const float*)d_in[2];
  const float* qkv_b  = (const float*)d_in[3];
  const float* proj_w = (const float*)d_in[4];
  const float* proj_b = (const float*)d_in[5];
  const float* btab   = (const float*)d_in[6];
  const int*   ridx   = (const int*)d_in[7];

  const int B_ = in_sizes[0] / (NTOK * DIM);   // 4096 windows
  const int nW = in_sizes[1] / (NTOK * NTOK);  // 1024 masks

  char* ws = (char*)d_ws;
  _Float16* wqkvH  = (_Float16*)(ws + WS_WQKVH);
  _Float16* wprojT = (_Float16*)(ws + WS_WPROJT);
  float*    biasF  = (float*)(ws + WS_BIASF);

  winattn_prep_kernel<<<128, 256, 0, stream>>>(qkv_w, proj_w, btab, ridx,
                                               wqkvH, wprojT, biasF);
  winattn_main_kernel<<<B_, 256, SMEM_HALVES * sizeof(_Float16), stream>>>(
      x, mask, qkv_b, proj_b, wqkvH, wprojT, biasF, (float*)d_out, nW);
}
